// CompressedModel_18262200942797
// MI455X (gfx1250) — matrix operand order repaired and checked by reading, not compiled, not run on hardware
//
#include <hip/hip_runtime.h>
#include <hip/hip_bf16.h>
#include <math.h>

// Problem constants (from reference: x is (4, 4096, 256) f32, R_RATIO=0.95)
#define B_    4
#define T_    4096
#define C_    256
#define R_    204              // floor(T - 0.95*T)
#define PROT_ (T_ - 2 * R_)    // 3688 protected tokens
#define OUTT_ (PROT_ + R_)     // 3892 output tokens per batch
#define EPS_  1e-12f
#define WPB   4                // waves per block in the WMMA passes
#define TILE_BYTES 8192        // one staged tile: 16 rows x 512 B, contiguous

typedef __attribute__((ext_vector_type(16))) __bf16       v16bf;
typedef __attribute__((ext_vector_type(8)))  float        v8f;
typedef __attribute__((ext_vector_type(4)))  unsigned int v4u;
typedef __attribute__((ext_vector_type(8)))  int          v8i;
typedef __attribute__((ext_vector_type(4)))  int          v4i;

union FragU { v4u q[2]; v16bf v; };

static __device__ __forceinline__ v8f wmma_bf16(v16bf a, v16bf b, v8f c) {
  // D = A(16x32 bf16) * B(32x16 bf16) + C(16x16 f32)
  return __builtin_amdgcn_wmma_f32_16x16x32_bf16(false, a, false, b, (short)0, c, false, false);
}

// ---------------------------------------------------------------------------
// TDM: 1-D global->LDS DMA of `nbytes` contiguous bytes (multiple of 8).
// D# packing per CDNA5 ISA:
//   group0: [1:0]=count=1 | [63:32]=lds_addr | [120:64]=global_addr | [127:126]=type=2
//   group1: [17:16]=data_size=3 (8B units) | tensor_dim0=tile_dim0=n |
//           tensor_dim1=1 | tensor_dim0_stride=n | tile_dim1/2=0 (1-D)
//   groups 2/3 (+trailing group): zero (<=2-D tensor; VADDR2/3 disabled)
// This toolchain exposes the 6-arg builtin:
//   (uint32x4, int32x8, int32x4, int32x4, int32x8, i32 cpol)
// Tracked by TENSORcnt; waited with s_wait_tensorcnt. EXEC is ignored by TDM,
// so only one wave per block issues these.
// ---------------------------------------------------------------------------
static __device__ __forceinline__ void tdm_load_1d(const void* gsrc, void* lds, int nbytes) {
  unsigned long long ga = (unsigned long long)(size_t)gsrc;
  unsigned ldsOff = (unsigned)(size_t)lds;      // flat LDS addr: low 32 bits = LDS offset
  unsigned n = (unsigned)(nbytes >> 3);         // 8-byte units (fits tile_dim0's 16 bits)
  v4u g0;
  g0.x = 1u;                                          // count=1, user descriptor
  g0.y = ldsOff;                                      // lds_addr
  g0.z = (unsigned)ga;                                // global_addr[31:0]
  g0.w = (unsigned)((ga >> 32) & 0x01FFFFFFu) | (2u << 30);  // addr[56:32] | type=2
  v8i g1;
  g1[0] = (3 << 16);                                  // workgroup_mask=0, data_size=8B
  g1[1] = (int)((n & 0xFFFFu) << 16);                 // tensor_dim0[15:0]
  g1[2] = (int)((n >> 16) | (1u << 16));              // tensor_dim0[31:16], tensor_dim1=1
  g1[3] = (int)((n & 0xFFFFu) << 16);                 // tile_dim0 = n
  g1[4] = 0;                                          // tile_dim1=0 (1-D), tile_dim2=0
  g1[5] = (int)n;                                     // tensor_dim0_stride[31:0]
  g1[6] = 0;
  g1[7] = 0;
  v4i z4 = {0, 0, 0, 0};
  v8i z8 = {0, 0, 0, 0, 0, 0, 0, 0};
  __builtin_amdgcn_tensor_load_to_lds(g0, g1, z4, z4, z8, 0);
}

// ---------------------------------------------------------------------------
// K0: row-normalize x, emit compensated bf16 split xn = hi + lo
// ---------------------------------------------------------------------------
__global__ __launch_bounds__(256) void k_normalize(const float* __restrict__ x,
                                                   __bf16* __restrict__ hi,
                                                   __bf16* __restrict__ lo) {
  int row  = blockIdx.x * 8 + (threadIdx.x >> 5);
  int lane = threadIdx.x & 31;
  const float* xr = x + (size_t)row * C_;
  float v[8];
  float ss = 0.f;
#pragma unroll
  for (int i = 0; i < 8; ++i) { v[i] = xr[lane * 8 + i]; ss += v[i] * v[i]; }
#pragma unroll
  for (int m = 1; m < 32; m <<= 1) ss += __shfl_xor(ss, m, 32);
  float scale = 1.0f / fmaxf(sqrtf(ss), EPS_);
  __bf16* hr = hi + (size_t)row * C_;
  __bf16* lr = lo + (size_t)row * C_;
#pragma unroll
  for (int i = 0; i < 8; ++i) {
    float  xs  = v[i] * scale;
    __bf16 h   = (__bf16)xs;
    float  rem = xs - (float)h;
    hr[lane * 8 + i] = h;
    lr[lane * 8 + i] = (__bf16)rem;
  }
}

// ---------------------------------------------------------------------------
// K1 (flash pass 1): denr[t] = 1 / sum_s exp(-sim[t,s])
// Wave owns a 16-row t-tile (A frags resident, hi+lo, 8 K-chunks = 128 VGPRs).
// s-tiles stream through double-buffered LDS via TDM (wave 0 issues; the
// s_wait_tensorcnt + workgroup barrier publishes the buffer to all waves).
// sim in [-1,1] => softmax max-subtraction skipped exactly-safely.
// ---------------------------------------------------------------------------
__global__ __launch_bounds__(WPB * 32) void k_denom(const __bf16* __restrict__ hi,
                                                    const __bf16* __restrict__ lo,
                                                    float* __restrict__ denr) {
  __shared__ v4u sHi[2][512];   // 2 x 8KB
  __shared__ v4u sLo[2][512];
  int tid  = threadIdx.x;
  int wave = tid >> 5, lane = tid & 31;
  int tileBase = blockIdx.x * WPB;
  int b  = tileBase >> 8;                      // 256 tiles per batch, WPB | 256
  int tb = ((tileBase + wave) & 255) * 16;     // row-tile base within batch
  int m = lane & 15, hh = lane >> 4;

  // Owned A fragments (rows tb..tb+15), hi and lo, all 8 K-chunks.
  v16bf a_hi[8], a_lo[8];
  {
    const char* bh = (const char*)hi + ((size_t)(b * T_ + tb + m)) * (C_ * 2);
    const char* bl = (const char*)lo + ((size_t)(b * T_ + tb + m)) * (C_ * 2);
#pragma unroll
    for (int kc = 0; kc < 8; ++kc) {
      FragU f;
      f.q[0] = *(const v4u*)(bh + kc * 64 + hh * 16);
      f.q[1] = *(const v4u*)(bh + kc * 64 + 32 + hh * 16);
      a_hi[kc] = f.v;
      f.q[0] = *(const v4u*)(bl + kc * 64 + hh * 16);
      f.q[1] = *(const v4u*)(bl + kc * 64 + 32 + hh * 16);
      a_lo[kc] = f.v;
    }
  }

  const char* gH = (const char*)hi + (size_t)b * T_ * (C_ * 2);
  const char* gL = (const char*)lo + (size_t)b * T_ * (C_ * 2);

  if (wave == 0) {                              // prime buffer 0
    tdm_load_1d(gH, &sHi[0][0], TILE_BYTES);
    tdm_load_1d(gL, &sLo[0][0], TILE_BYTES);
    __builtin_amdgcn_s_wait_tensorcnt(0);
  }
  __syncthreads();

  float racc[8] = {0, 0, 0, 0, 0, 0, 0, 0};

  for (int st = 0; st < T_ / 16; ++st) {
    int cur = st & 1;
    if (wave == 0 && st + 1 < T_ / 16) {        // async-prefetch next tile
      size_t off = (size_t)(st + 1) * TILE_BYTES;
      tdm_load_1d(gH + off, &sHi[cur ^ 1][0], TILE_BYTES);
      tdm_load_1d(gL + off, &sLo[cur ^ 1][0], TILE_BYTES);
    }

    // Two independent accumulator chains: hi*hi and (hi*lo + lo*hi)
    v8f accH = {0.f, 0.f, 0.f, 0.f, 0.f, 0.f, 0.f, 0.f};
    v8f accM = {0.f, 0.f, 0.f, 0.f, 0.f, 0.f, 0.f, 0.f};
#pragma unroll
    for (int kc = 0; kc < 8; ++kc) {
      int bi = m * 32 + kc * 4 + hh;            // B-matrix lane layout, v4u units
      FragU bhf, blf;
      bhf.q[0] = sHi[cur][bi];
      bhf.q[1] = sHi[cur][bi + 2];
      blf.q[0] = sLo[cur][bi];
      blf.q[1] = sLo[cur][bi + 2];
      accM = wmma_bf16(a_lo[kc], bhf.v, accM);
      accM = wmma_bf16(a_hi[kc], blf.v, accM);
      accH = wmma_bf16(a_hi[kc], bhf.v, accH);
    }
#pragma unroll
    for (int j = 0; j < 8; ++j) racc[j] += __expf(-(accH[j] + accM[j]));

    if (wave == 0 && st + 1 < T_ / 16) __builtin_amdgcn_s_wait_tensorcnt(0);
    __syncthreads();
  }

  // Row sums: reduce across the 16 lanes of each half (rows j + 8*hh).
#pragma unroll
  for (int j = 0; j < 8; ++j) {
    float rs = racc[j];
    rs += __shfl_xor(rs, 1, 32);
    rs += __shfl_xor(rs, 2, 32);
    rs += __shfl_xor(rs, 4, 32);
    rs += __shfl_xor(rs, 8, 32);
    if (m == 0) denr[b * T_ + tb + j + 8 * hh] = 1.0f / rs;
  }
}

// ---------------------------------------------------------------------------
// K2 (flash pass 2): iso[s] = (1/T) * sum_t exp(-sim[t,s]) * denr[t]
// Roles swapped: wave owns a 16-column s-tile (B frags resident), t-tiles
// stream through double-buffered LDS via TDM.
// ---------------------------------------------------------------------------
__global__ __launch_bounds__(WPB * 32) void k_iso(const __bf16* __restrict__ hi,
                                                  const __bf16* __restrict__ lo,
                                                  const float* __restrict__ denr,
                                                  float* __restrict__ iso) {
  __shared__ v4u sHi[2][512];
  __shared__ v4u sLo[2][512];
  int tid  = threadIdx.x;
  int wave = tid >> 5, lane = tid & 31;
  int tileBase = blockIdx.x * WPB;
  int b  = tileBase >> 8;
  int sb = ((tileBase + wave) & 255) * 16;     // owned column tile
  int n = lane & 15, hh = lane >> 4;

  v16bf b_hi[8], b_lo[8];
  {
    const char* bh = (const char*)hi + ((size_t)(b * T_ + sb + n)) * (C_ * 2);
    const char* bl = (const char*)lo + ((size_t)(b * T_ + sb + n)) * (C_ * 2);
#pragma unroll
    for (int kc = 0; kc < 8; ++kc) {
      FragU f;
      f.q[0] = *(const v4u*)(bh + kc * 64 + hh * 16);
      f.q[1] = *(const v4u*)(bh + kc * 64 + 32 + hh * 16);
      b_hi[kc] = f.v;
      f.q[0] = *(const v4u*)(bl + kc * 64 + hh * 16);
      f.q[1] = *(const v4u*)(bl + kc * 64 + 32 + hh * 16);
      b_lo[kc] = f.v;
    }
  }

  const char* gH = (const char*)hi + (size_t)b * T_ * (C_ * 2);
  const char* gL = (const char*)lo + (size_t)b * T_ * (C_ * 2);

  if (wave == 0) {
    tdm_load_1d(gH, &sHi[0][0], TILE_BYTES);
    tdm_load_1d(gL, &sLo[0][0], TILE_BYTES);
    __builtin_amdgcn_s_wait_tensorcnt(0);
  }
  __syncthreads();

  float colacc = 0.f;

  for (int tt = 0; tt < T_ / 16; ++tt) {
    int cur = tt & 1;
    if (wave == 0 && tt + 1 < T_ / 16) {
      size_t off = (size_t)(tt + 1) * TILE_BYTES;
      tdm_load_1d(gH + off, &sHi[cur ^ 1][0], TILE_BYTES);
      tdm_load_1d(gL + off, &sLo[cur ^ 1][0], TILE_BYTES);
    }

    v8f accH = {0.f, 0.f, 0.f, 0.f, 0.f, 0.f, 0.f, 0.f};
    v8f accM = {0.f, 0.f, 0.f, 0.f, 0.f, 0.f, 0.f, 0.f};
#pragma unroll
    for (int kc = 0; kc < 8; ++kc) {
      int ai = n * 32 + kc * 4;                 // A-matrix lane layout, v4u units
      FragU ah, al;
      ah.q[0] = sHi[cur][ai + hh];
      ah.q[1] = sHi[cur][ai + 2 + hh];
      al.q[0] = sLo[cur][ai + hh];
      al.q[1] = sLo[cur][ai + 2 + hh];
      accM = wmma_bf16(al.v, b_hi[kc], accM);
      accM = wmma_bf16(ah.v, b_lo[kc], accM);
      accH = wmma_bf16(ah.v, b_hi[kc], accH);
    }
    int tbase = tt * 16;
#pragma unroll
    for (int j = 0; j < 8; ++j) {
      float rd = denr[b * T_ + tbase + j + 8 * hh];
      colacc += __expf(-(accH[j] + accM[j])) * rd;
    }

    if (wave == 0 && tt + 1 < T_ / 16) __builtin_amdgcn_s_wait_tensorcnt(0);
    __syncthreads();
  }
  colacc += __shfl_xor(colacc, 16, 32);        // fold the two row-halves
  if (lane < 16) iso[b * T_ + sb + lane] = colacc * (1.0f / T_);
}

// ---------------------------------------------------------------------------
// K3: deterministic argsort via rank counting. idx[b][rank] = token
// ---------------------------------------------------------------------------
__global__ __launch_bounds__(256) void k_rank(const float* __restrict__ iso,
                                              int* __restrict__ idx) {
  int b = blockIdx.y;
  int t = blockIdx.x * 256 + threadIdx.x;
  const float* p = iso + b * T_;
  float v = p[t];
  int cnt = 0;
  for (int s = 0; s < T_; ++s) {
    float u = p[s];
    cnt += (u < v) || (u == v && s < t);       // stable ascending, like jnp.argsort
  }
  idx[b * T_ + cnt] = t;
}

// ---------------------------------------------------------------------------
// K4: dst_idx[i] = argmax_j sim[a_i, b_j]  (first max wins, like jnp.argmax)
// ---------------------------------------------------------------------------
__global__ __launch_bounds__(128) void k_dst(const __bf16* __restrict__ hi,
                                             const __bf16* __restrict__ lo,
                                             const int* __restrict__ idx,
                                             int* __restrict__ dstidx) {
  __shared__ float arow[C_];
  __shared__ float sc[R_];
  int b = blockIdx.y, i = blockIdx.x, tid = threadIdx.x;
  int ai = idx[b * T_ + 2 * i];
  const __bf16* hr = hi + (size_t)(b * T_ + ai) * C_;
  const __bf16* lr = lo + (size_t)(b * T_ + ai) * C_;
  for (int c = tid; c < C_; c += 128) arow[c] = (float)hr[c] + (float)lr[c];
  __syncthreads();
  for (int j = tid; j < R_; j += 128) {
    int bj = idx[b * T_ + 2 * j + 1];
    const __bf16* hb = hi + (size_t)(b * T_ + bj) * C_;
    const __bf16* lb = lo + (size_t)(b * T_ + bj) * C_;
    float s = 0.f;
    for (int c = 0; c < C_; ++c) s += arow[c] * ((float)hb[c] + (float)lb[c]);
    sc[j] = s;
  }
  __syncthreads();
  if (tid == 0) {
    int best = 0; float bv = sc[0];
    for (int j = 1; j < R_; ++j) if (sc[j] > bv) { bv = sc[j]; best = j; }
    dstidx[b * R_ + i] = best;
  }
}

// ---------------------------------------------------------------------------
// K5: protected rows — xm/sm = (x*iso)/iso = x, so a straight gather copy
// ---------------------------------------------------------------------------
__global__ __launch_bounds__(256) void k_prot(const float* __restrict__ x,
                                              const int* __restrict__ idx,
                                              float* __restrict__ out) {
  int b = blockIdx.y, row = blockIdx.x, c = threadIdx.x;
  int p = idx[b * T_ + 2 * R_ + row];
  out[((size_t)b * OUTT_ + row) * C_ + c] = x[((size_t)b * T_ + p) * C_ + c];
}

// ---------------------------------------------------------------------------
// K6: dst rows — weighted merge; scan-style (no atomics -> bitwise
// deterministic across graph replays)
// ---------------------------------------------------------------------------
__global__ __launch_bounds__(256) void k_merge(const float* __restrict__ x,
                                               const float* __restrict__ iso,
                                               const int* __restrict__ idx,
                                               const int* __restrict__ dstidx,
                                               float* __restrict__ out) {
  int b = blockIdx.y, j = blockIdx.x, c = threadIdx.x;
  int bj = idx[b * T_ + 2 * j + 1];
  float wj  = iso[b * T_ + bj];
  float num = x[((size_t)b * T_ + bj) * C_ + c] * wj;
  float den = wj;
  for (int i = 0; i < R_; ++i) {
    if (dstidx[b * R_ + i] == j) {
      int ai = idx[b * T_ + 2 * i];
      float wa = iso[b * T_ + ai];
      num += x[((size_t)b * T_ + ai) * C_ + c] * wa;
      den += wa;
    }
  }
  out[((size_t)b * OUTT_ + PROT_ + j) * C_ + c] = num / den;
}

// ---------------------------------------------------------------------------
// Launcher. Workspace layout (bytes):
//   [0,        8M)  xn_hi  (bf16, B*T*C)
//   [8M,      16M)  xn_lo  (bf16, B*T*C)
//   [16M,  +64K)    denr   (f32, B*T)  reciprocal softmax denominators
//   [+64K, +128K)   iso    (f32, B*T)
//   [+128K,+192K)   idx    (i32, B*T)  argsort of iso
//   [+192K,+196K)   dstidx (i32, B*R)
// Total ~16.2 MB.
// ---------------------------------------------------------------------------
extern "C" void kernel_launch(void* const* d_in, const int* in_sizes, int n_in,
                              void* d_out, int out_size, void* d_ws, size_t ws_size,
                              hipStream_t stream) {
  (void)in_sizes; (void)n_in; (void)out_size; (void)ws_size;
  const float* x = (const float*)d_in[0];
  float* out = (float*)d_out;
  char* ws = (char*)d_ws;

  const size_t HALF = (size_t)B_ * T_ * C_ * 2;  // 8 MiB
  __bf16* hi   = (__bf16*)(ws);
  __bf16* lo   = (__bf16*)(ws + HALF);
  float* denr  = (float*)(ws + 2 * HALF);
  float* iso   = (float*)(ws + 2 * HALF + 64 * 1024);
  int*   idx   = (int*)  (ws + 2 * HALF + 128 * 1024);
  int*   dsti  = (int*)  (ws + 2 * HALF + 192 * 1024);

  k_normalize<<<dim3(B_ * T_ / 8), dim3(256), 0, stream>>>(x, hi, lo);
  k_denom<<<dim3(B_ * T_ / 16 / WPB), dim3(WPB * 32), 0, stream>>>(hi, lo, denr);
  k_iso<<<dim3(B_ * T_ / 16 / WPB), dim3(WPB * 32), 0, stream>>>(hi, lo, denr, iso);
  k_rank<<<dim3(T_ / 256, B_), dim3(256), 0, stream>>>(iso, idx);
  k_dst<<<dim3(R_, B_), dim3(128), 0, stream>>>(hi, lo, idx, dsti);
  k_prot<<<dim3(PROT_, B_), dim3(256), 0, stream>>>(x, idx, out);
  k_merge<<<dim3(R_, B_), dim3(256), 0, stream>>>(x, iso, idx, dsti, out);
}
